// LocalSorterModel_84456236908736
// MI455X (gfx1250) — compile-verified, matrix operand order checked by reading, not run on hardware
//
#include <hip/hip_runtime.h>

// ---------------- problem constants ----------------
#define BATCH   2048
#define DIM     1024
#define NITEMS  5
#define NRES    120
#define KX      (NITEMS * DIM)   // 5120  : contraction for main GEMM
#define MCW     (NRES * NITEMS)  // 600   : rows of CWcat / M1
#define KCW     (2 * DIM)        // 2048  : contraction for M1 GEMM

typedef __attribute__((ext_vector_type(8)))  __bf16 v8bf;
typedef __attribute__((ext_vector_type(16))) __bf16 v16bf;
typedef __attribute__((ext_vector_type(8)))  float  v8f;

// ---------------- helpers ----------------

// fp32 -> (hi, lo) bf16 pair; hi+lo reproduces x to ~2^-16 relative
__device__ __forceinline__ void split2(float x, __bf16* h, __bf16* l) {
  __bf16 hi = (__bf16)x;
  *h = hi;
  *l = (__bf16)(x - (float)hi);
}

// Load one WMMA 16x16x32 bf16 operand fragment for this lane.
// p points at (row_or_col_base + half*8) in a K-contiguous array.
// Elements 0..7  -> VGPR0-3 : K = base .. base+7
// Elements 8..15 -> VGPR4-7 : K = base+16 .. base+23
__device__ __forceinline__ v16bf ldfrag(const __bf16* __restrict__ p) {
  v8bf a = *(const v8bf*)(p);
  v8bf b = *(const v8bf*)(p + 16);
  v16bf r;
#pragma unroll
  for (int t = 0; t < 8; ++t) { r[t] = a[t]; r[t + 8] = b[t]; }
  return r;
}

__device__ __forceinline__ v8f wmma_bf16(v16bf a, v16bf b, v8f c) {
  return __builtin_amdgcn_wmma_f32_16x16x32_bf16(false, a, false, b, (short)0, c,
                                                 false, false);
}

// bf16x3, 2x2 register-blocked K-loop: each wave owns a 32x32 output tile.
// Per K=32 step: 8 fragment loads (16 x b128) feed 12 WMMAs.
// A = Ah+Al, B = Bh+Bl, product approximated hi*hi + hi*lo + lo*hi (fp32 acc).
__device__ __forceinline__ void gemm_2x2(const __bf16* __restrict__ ah0,
                                         const __bf16* __restrict__ al0,
                                         const __bf16* __restrict__ ah1,
                                         const __bf16* __restrict__ al1,
                                         const __bf16* __restrict__ bh0,
                                         const __bf16* __restrict__ bl0,
                                         const __bf16* __restrict__ bh1,
                                         const __bf16* __restrict__ bl1,
                                         int K,
                                         v8f& c00, v8f& c01, v8f& c10, v8f& c11) {
#pragma unroll 2
  for (int kb = 0; kb < K; kb += 32) {
    v16bf fa0h = ldfrag(ah0 + kb), fa0l = ldfrag(al0 + kb);
    v16bf fa1h = ldfrag(ah1 + kb), fa1l = ldfrag(al1 + kb);
    v16bf fb0h = ldfrag(bh0 + kb), fb0l = ldfrag(bl0 + kb);
    v16bf fb1h = ldfrag(bh1 + kb), fb1l = ldfrag(bl1 + kb);

    c00 = wmma_bf16(fa0h, fb0h, c00);
    c01 = wmma_bf16(fa0h, fb1h, c01);
    c10 = wmma_bf16(fa1h, fb0h, c10);
    c11 = wmma_bf16(fa1h, fb1h, c11);

    c00 = wmma_bf16(fa0h, fb0l, c00);
    c01 = wmma_bf16(fa0h, fb1l, c01);
    c10 = wmma_bf16(fa1h, fb0l, c10);
    c11 = wmma_bf16(fa1h, fb1l, c11);

    c00 = wmma_bf16(fa0l, fb0h, c00);
    c01 = wmma_bf16(fa0l, fb1h, c01);
    c10 = wmma_bf16(fa1l, fb0h, c10);
    c11 = wmma_bf16(fa1l, fb1h, c11);
  }
}

// ---------------- prep kernels ----------------

// embeds (2048 x 5120 f32, already flat) -> Xh/Xl bf16 split
__global__ __launch_bounds__(256) void prep_x(const float* __restrict__ e,
                                              __bf16* __restrict__ Xh,
                                              __bf16* __restrict__ Xl) {
  long i = (long)blockIdx.x * blockDim.x + threadIdx.x;
  const long n4 = (long)BATCH * KX / 4;
  if (i >= n4) return;
  float4 v = ((const float4*)e)[i];
  long o = i * 4;
  split2(v.x, &Xh[o + 0], &Xl[o + 0]);
  split2(v.y, &Xh[o + 1], &Xl[o + 1]);
  split2(v.z, &Xh[o + 2], &Xl[o + 2]);
  split2(v.w, &Xh[o + 3], &Xl[o + 3]);
}

// CWcat[(r*5+i)][k'] : k'<1024 -> sum of cls_w over pairs with I==i
//                      k'>=1024 -> sum over pairs with J==i
__global__ __launch_bounds__(256) void prep_cw(const float* __restrict__ cls_w,
                                               __bf16* __restrict__ CWh,
                                               __bf16* __restrict__ CWl) {
  long idx = (long)blockIdx.x * blockDim.x + threadIdx.x;
  if (idx >= (long)MCW * KCW) return;
  int kp = (int)(idx & (KCW - 1));
  int m  = (int)(idx >> 11);
  int r = m / NITEMS, i = m % NITEMS;
  float s = 0.f;
  if (kp < DIM) {
    int k = kp;
#pragma unroll
    for (int t = 0; t < 4; ++t)
      s += cls_w[(long)r * 20480 + (4 * i + t) * DIM + k];
  } else {
    int k = kp - DIM;
#pragma unroll
    for (int a = 0; a < NITEMS; ++a) {
      if (a == i) continue;
      int p = 4 * a + i - (i > a ? 1 : 0);   // pair with (I=a, J=i)
      s += cls_w[(long)r * 20480 + p * DIM + k];
    }
  }
  split2(s, &CWh[idx], &CWl[idx]);
}

// WcatT[d][k'] = pw_w[k'%1024][ d + (k'<1024 ? 0 : 1024) ]  (N-major so B frags are K-contiguous)
__global__ __launch_bounds__(256) void prep_w(const float* __restrict__ pw_w,
                                              __bf16* __restrict__ Wh,
                                              __bf16* __restrict__ Wl) {
  long idx = (long)blockIdx.x * blockDim.x + threadIdx.x;
  if (idx >= (long)DIM * KCW) return;
  int kp = (int)(idx & (KCW - 1));
  int d  = (int)(idx >> 11);
  int k  = kp & (DIM - 1);
  int col = (kp < DIM) ? d : (d + DIM);
  float v = pw_w[(long)k * KCW + col];
  split2(v, &Wh[idx], &Wl[idx]);
}

// bias[r] = cls_b[r] + sum_t cls_w[r][t] * pw_b[t % 1024]
__global__ __launch_bounds__(256) void bias_k(const float* __restrict__ cls_w,
                                              const float* __restrict__ cls_b,
                                              const float* __restrict__ pw_b,
                                              float* __restrict__ bias) {
  __shared__ float red[256];
  int r = blockIdx.x;
  float s = 0.f;
  for (int t = threadIdx.x; t < 20480; t += 256)
    s += cls_w[(long)r * 20480 + t] * pw_b[t & (DIM - 1)];
  red[threadIdx.x] = s;
  __syncthreads();
  for (int o = 128; o > 0; o >>= 1) {
    if (threadIdx.x < (unsigned)o) red[threadIdx.x] += red[threadIdx.x + o];
    __syncthreads();
  }
  if (threadIdx.x == 0) bias[r] = red[0] + cls_b[r];
}

// ---------------- GEMM 1: M1 = CWcat @ Wcat   (600 x 2048 x 1024) ----------------
// 2x2 blocked: 19 M-supertiles (32 rows) x 32 N-supertiles (32 cols) = 608 waves.
// writes M1 split into bf16 hi/lo, stored [600][1024] == [120][5120] (r outer)
__global__ __launch_bounds__(256) void gemm1(const __bf16* __restrict__ CWh,
                                             const __bf16* __restrict__ CWl,
                                             const __bf16* __restrict__ Wh,
                                             const __bf16* __restrict__ Wl,
                                             __bf16* __restrict__ Mh,
                                             __bf16* __restrict__ Ml) {
  int wave = blockIdx.x * 8 + (threadIdx.x >> 5);
  int nst  = wave & 31;           // 32 supertiles over N=1024
  int mst  = wave >> 5;           // 19 supertiles over M=600 (padded to 608)
  int lane = threadIdx.x & 31;
  int half = lane >> 4;
  int lr   = lane & 15;

  int ar0 = mst * 32 + lr;      if (ar0 > MCW - 1) ar0 = MCW - 1;
  int ar1 = mst * 32 + 16 + lr; if (ar1 > MCW - 1) ar1 = MCW - 1;
  int bc0 = nst * 32 + lr;
  int bc1 = nst * 32 + 16 + lr;

  const __bf16* ah0 = CWh + (long)ar0 * KCW + half * 8;
  const __bf16* al0 = CWl + (long)ar0 * KCW + half * 8;
  const __bf16* ah1 = CWh + (long)ar1 * KCW + half * 8;
  const __bf16* al1 = CWl + (long)ar1 * KCW + half * 8;
  const __bf16* bh0 = Wh  + (long)bc0 * KCW + half * 8;
  const __bf16* bl0 = Wl  + (long)bc0 * KCW + half * 8;
  const __bf16* bh1 = Wh  + (long)bc1 * KCW + half * 8;
  const __bf16* bl1 = Wl  + (long)bc1 * KCW + half * 8;

  v8f c00 = {0.f,0.f,0.f,0.f,0.f,0.f,0.f,0.f};
  v8f c01 = c00, c10 = c00, c11 = c00;
  gemm_2x2(ah0, al0, ah1, al1, bh0, bl0, bh1, bl1, KCW, c00, c01, c10, c11);

  int n0 = nst * 32 + lr;
  int n1 = n0 + 16;
#pragma unroll
  for (int r2 = 0; r2 < 8; ++r2) {
    int m0 = mst * 32 + r2 + half * 8;
    int m1 = m0 + 16;
    if (m0 < MCW) {
      split2(c00[r2], &Mh[(long)m0 * DIM + n0], &Ml[(long)m0 * DIM + n0]);
      split2(c01[r2], &Mh[(long)m0 * DIM + n1], &Ml[(long)m0 * DIM + n1]);
    }
    if (m1 < MCW) {
      split2(c10[r2], &Mh[(long)m1 * DIM + n0], &Ml[(long)m1 * DIM + n0]);
      split2(c11[r2], &Mh[(long)m1 * DIM + n1], &Ml[(long)m1 * DIM + n1]);
    }
  }
}

// ---------------- GEMM 2: out = X @ M1^T + bias   (2048 x 5120 x 120) ----------------
// 2x2 blocked: 64 M-supertiles x 4 N-supertiles = 256 waves.
__global__ __launch_bounds__(256) void gemm2(const __bf16* __restrict__ Xh,
                                             const __bf16* __restrict__ Xl,
                                             const __bf16* __restrict__ Mh,
                                             const __bf16* __restrict__ Ml,
                                             const float* __restrict__ bias,
                                             float* __restrict__ out) {
  int wave = blockIdx.x * 8 + (threadIdx.x >> 5);
  int nst  = wave & 3;            // 4 supertiles over N=120 (padded to 128)
  int mst  = wave >> 2;           // 64 supertiles over M=2048
  int lane = threadIdx.x & 31;
  int half = lane >> 4;
  int lr   = lane & 15;

  int ar0 = mst * 32 + lr;        // < 2048
  int ar1 = ar0 + 16;
  int bc0 = nst * 32 + lr;      if (bc0 > NRES - 1) bc0 = NRES - 1;
  int bc1 = nst * 32 + 16 + lr; if (bc1 > NRES - 1) bc1 = NRES - 1;

  const __bf16* ah0 = Xh + (long)ar0 * KX + half * 8;
  const __bf16* al0 = Xl + (long)ar0 * KX + half * 8;
  const __bf16* ah1 = Xh + (long)ar1 * KX + half * 8;
  const __bf16* al1 = Xl + (long)ar1 * KX + half * 8;
  const __bf16* bh0 = Mh + (long)bc0 * KX + half * 8;   // M1 row r == M1^T column r
  const __bf16* bl0 = Ml + (long)bc0 * KX + half * 8;
  const __bf16* bh1 = Mh + (long)bc1 * KX + half * 8;
  const __bf16* bl1 = Ml + (long)bc1 * KX + half * 8;

  v8f c00 = {0.f,0.f,0.f,0.f,0.f,0.f,0.f,0.f};
  v8f c01 = c00, c10 = c00, c11 = c00;
  gemm_2x2(ah0, al0, ah1, al1, bh0, bl0, bh1, bl1, KX, c00, c01, c10, c11);

  int n0 = nst * 32 + lr;
  int n1 = n0 + 16;
  float bv0 = (n0 < NRES) ? bias[n0] : 0.f;
  float bv1 = (n1 < NRES) ? bias[n1] : 0.f;
#pragma unroll
  for (int r2 = 0; r2 < 8; ++r2) {
    int m0 = mst * 32 + r2 + half * 8;
    int m1 = m0 + 16;
    if (n0 < NRES) {
      out[(long)m0 * NRES + n0] = c00[r2] + bv0;
      out[(long)m1 * NRES + n0] = c10[r2] + bv0;
    }
    if (n1 < NRES) {
      out[(long)m0 * NRES + n1] = c01[r2] + bv1;
      out[(long)m1 * NRES + n1] = c11[r2] + bv1;
    }
  }
}

// ---------------- launcher ----------------
extern "C" void kernel_launch(void* const* d_in, const int* in_sizes, int n_in,
                              void* d_out, int out_size, void* d_ws, size_t ws_size,
                              hipStream_t stream) {
  const float* embeds = (const float*)d_in[0];   // (2048, 5, 1024)
  const float* pw_w   = (const float*)d_in[1];   // (1024, 2048)
  const float* pw_b   = (const float*)d_in[2];   // (1024,)
  const float* cls_w  = (const float*)d_in[3];   // (120, 20480)
  const float* cls_b  = (const float*)d_in[4];   // (120,)
  float* out = (float*)d_out;                    // (2048, 120)

  // workspace carve-up (bf16 elements, then fp32 bias)
  __bf16* base = (__bf16*)d_ws;
  size_t off = 0;
  __bf16* Xh  = base + off; off += (size_t)BATCH * KX;   // 10,485,760
  __bf16* Xl  = base + off; off += (size_t)BATCH * KX;
  __bf16* CWh = base + off; off += (size_t)MCW * KCW;    // 1,228,800
  __bf16* CWl = base + off; off += (size_t)MCW * KCW;
  __bf16* Wh  = base + off; off += (size_t)DIM * KCW;    // 2,097,152
  __bf16* Wl  = base + off; off += (size_t)DIM * KCW;
  __bf16* Mh  = base + off; off += (size_t)MCW * DIM;    // 614,400
  __bf16* Ml  = base + off; off += (size_t)MCW * DIM;
  float* bias = (float*)(base + off);                    // 480 B, 4B-aligned

  // prep
  prep_x <<<(int)(((long)BATCH * KX / 4 + 255) / 256), 256, 0, stream>>>(embeds, Xh, Xl);
  prep_cw<<<(int)(((long)MCW * KCW + 255) / 256),      256, 0, stream>>>(cls_w, CWh, CWl);
  prep_w <<<(int)(((long)DIM * KCW + 255) / 256),      256, 0, stream>>>(pw_w, Wh, Wl);
  bias_k <<<NRES, 256, 0, stream>>>(cls_w, cls_b, pw_b, bias);

  // GEMM1: 19 x 32 supertiles (32x32 per wave), 8 waves/block -> 76 blocks
  gemm1<<<76, 256, 0, stream>>>(CWh, CWl, Wh, Wl, Mh, Ml);

  // GEMM2: 64 x 4 supertiles (32x32 per wave), 8 waves/block -> 32 blocks
  gemm2<<<32, 256, 0, stream>>>(Xh, Xl, Mh, Ml, bias, out);
}
